// MSA1_7421703487716
// MI455X (gfx1250) — matrix operand order repaired and checked by reading, not compiled, not run on hardware
//
#include <hip/hip_runtime.h>

#define BATCH 4
#define LSEQ  2048
#define NHID_ 256
#define NH    10
#define HD    10
#define DTOT  30   // HD * (1 + AL)
#define DPAD  32

typedef __attribute__((ext_vector_type(16))) __bf16 v16bf;
typedef __attribute__((ext_vector_type(8)))  __bf16 v8bf;
typedef __attribute__((ext_vector_type(8)))  float  v8f;

static __device__ __forceinline__ v16bf cat8(v8bf lo, v8bf hi) {
  return __builtin_shufflevector(lo, hi, 0,1,2,3,4,5,6,7,8,9,10,11,12,13,14,15);
}

static __device__ __forceinline__ v8f wmma_bf16(v16bf a, v16bf b, v8f c) {
  // D = A(16x32 bf16) * B(32x16 bf16) + C(16x16 f32)
  return __builtin_amdgcn_wmma_f32_16x16x32_bf16(false, a, false, b, (short)0, c,
                                                 false, false);
}

// ---------------------------------------------------------------------------
// Kernel 1: QKV projection.  One block per (b,l).  Writes padded bf16 K/V in
// three layouts tailored to the WMMA fragment loads of the flash kernel.
// ---------------------------------------------------------------------------
__global__ __launch_bounds__(320) void msa_proj(
    const float* __restrict__ x,  const float* __restrict__ ax,
    const float* __restrict__ wq, const float* __restrict__ bq,
    const float* __restrict__ wk, const float* __restrict__ bk,
    const float* __restrict__ wv, const float* __restrict__ bv,
    __bf16* __restrict__ Kb, __bf16* __restrict__ Vb, __bf16* __restrict__ Vt)
{
  const int bl = blockIdx.x;
  const int b = bl >> 11, l = bl & (LSEQ - 1);
  __shared__ float xs[NHID_], a0[NHID_], a1[NHID_];
  const int tid = threadIdx.x;
  for (int c = tid; c < NHID_; c += 320) {
    xs[c] = x[((size_t)b * NHID_ + c) * LSEQ + l];
    a0[c] = ax[(((size_t)b * NHID_ + c) * 2 + 0) * LSEQ + l];
    a1[c] = ax[(((size_t)b * NHID_ + c) * 2 + 1) * LSEQ + l];
  }
  __syncthreads();

  const int h = tid >> 5, d = tid & 31;
  float vk = 0.f, vv = 0.f;
  if (d < HD) {                       // shared q row -> both K and V
    const int o = h * HD + d;
    const float* w = wq + o * NHID_;
    float s = bq[o];
    #pragma unroll 4
    for (int c = 0; c < NHID_; c += 4) {
      float4 w4 = *(const float4*)(w + c);
      float4 x4 = *(const float4*)(xs + c);
      s = fmaf(w4.x, x4.x, s); s = fmaf(w4.y, x4.y, s);
      s = fmaf(w4.z, x4.z, s); s = fmaf(w4.w, x4.w, s);
    }
    vk = s; vv = s;
  } else if (d < DTOT) {              // ak / av from ax slice 0 or 1
    const int slice = (d < 20) ? 0 : 1;
    const float* as = slice ? a1 : a0;
    const int o = h * HD + (d - 10) - slice * 10;
    const float* wkr = wk + o * NHID_;
    const float* wvr = wv + o * NHID_;
    float sk = bk[o], sv = bv[o];
    #pragma unroll 4
    for (int c = 0; c < NHID_; c += 4) {
      float4 a4 = *(const float4*)(as + c);
      float4 k4 = *(const float4*)(wkr + c);
      float4 v4 = *(const float4*)(wvr + c);
      sk = fmaf(k4.x, a4.x, sk); sk = fmaf(k4.y, a4.y, sk);
      sk = fmaf(k4.z, a4.z, sk); sk = fmaf(k4.w, a4.w, sk);
      sv = fmaf(v4.x, a4.x, sv); sv = fmaf(v4.y, a4.y, sv);
      sv = fmaf(v4.z, a4.z, sv); sv = fmaf(v4.w, a4.w, sv);
    }
    vk = sk; vv = sv;
  }
  // d == 30,31 fall through with zeros -> K-dim padding for the WMMA

  const size_t bh  = (size_t)(b * NH + h);
  const size_t idx = (bh * LSEQ + l) * DPAD + d;
  Kb[idx] = (__bf16)vk;
  Vb[idx] = (__bf16)vv;
  Vt[(bh * DPAD + d) * LSEQ + l] = (__bf16)vv;
}

// ---------------------------------------------------------------------------
// Kernel 2: flash attention.  Grid = B*NH*(L/64), 4 waves/block, each wave
// owns a 16-row l-strip.  S = K·Vt (1 WMMA covers full D=30->32 reduction),
// edge*scale + online softmax in f32, P routed through per-wave LDS to get
// the A-fragment interleave, O accumulated with 2 K=32 WMMA steps per m-tile.
// ---------------------------------------------------------------------------
__global__ __launch_bounds__(128) void msa_flash(
    const __bf16* __restrict__ Kb, const __bf16* __restrict__ Vb,
    const __bf16* __restrict__ Vt, const float* __restrict__ edge,
    float* __restrict__ attf)
{
  const int ntile = LSEQ / 64;
  const int bh    = blockIdx.x / ntile;
  const int ltile = (blockIdx.x % ntile) * 64;
  const int b     = bh / NH;
  const int lane  = threadIdx.x & 31;
  const int wave  = threadIdx.x >> 5;
  const int kh    = lane >> 4;     // which K-half of the A/B interleave
  const int ll    = lane & 15;

  __shared__ __bf16 P[4][16][64];  // per-wave bf16 probability tiles (8 KB)

  // Persistent A-fragment: K rows for this wave's 16 l-rows.
  // 16-bit A layout: lane(ll,kh) holds row M=ll, d in [8kh,8kh+8) u [16+8kh,..)
  const __bf16* Krow = Kb + ((size_t)bh * LSEQ + (ltile + wave * 16 + ll)) * DPAD;
  const v16bf aK = cat8(*(const v8bf*)(Krow + kh * 8),
                        *(const v8bf*)(Krow + 16 + kh * 8));

  v8f o0 = {0,0,0,0,0,0,0,0};
  v8f o1 = {0,0,0,0,0,0,0,0};
  float mrun[8], lrun[8];
  #pragma unroll
  for (int r = 0; r < 8; ++r) { mrun[r] = -__builtin_inff(); lrun[r] = 0.f; }

  const float* eB = edge + (size_t)b * LSEQ * LSEQ;
  const float scale = 0.31622776601683794f;   // 1/sqrt(HD)

  for (int mt = 0; mt < LSEQ; mt += 64) {
    // ---- S tile [16 x 64]: 4 WMMAs, B-frag = V rows (row-major, contiguous d)
    v8f s[4];
    #pragma unroll
    for (int t = 0; t < 4; ++t) {
      const __bf16* Vrow =
          Vb + ((size_t)bh * LSEQ + (mt + t * 16 + ll)) * DPAD;
      v16bf bf = cat8(*(const v8bf*)(Vrow + kh * 8), *(const v8bf*)(Vrow + 16 + kh * 8));
      v8f z = {0,0,0,0,0,0,0,0};
      s[t] = wmma_bf16(aK, bf, z);
    }
    // ---- elementwise edge * 1/sqrt(hd); edge is L2-resident across heads
    #pragma unroll
    for (int r = 0; r < 8; ++r) {
      const float* er =
          eB + (size_t)(ltile + wave * 16 + r + 8 * kh) * LSEQ + mt + ll;
      #pragma unroll
      for (int t = 0; t < 4; ++t)
        s[t][r] = s[t][r] * er[t * 16] * scale;
    }
    // ---- online softmax, one row per accumulator slot (M = r + 8*kh)
    #pragma unroll
    for (int r = 0; r < 8; ++r) {
      float tmax = fmaxf(fmaxf(s[0][r], s[1][r]), fmaxf(s[2][r], s[3][r]));
      #pragma unroll
      for (int off = 1; off <= 8; off <<= 1)
        tmax = fmaxf(tmax, __shfl_xor(tmax, off, 32));
      const float mnew = fmaxf(mrun[r], tmax);
      const float corr = __expf(mrun[r] - mnew);
      const float p0 = __expf(s[0][r] - mnew);
      const float p1 = __expf(s[1][r] - mnew);
      const float p2 = __expf(s[2][r] - mnew);
      const float p3 = __expf(s[3][r] - mnew);
      float ps = (p0 + p1) + (p2 + p3);
      #pragma unroll
      for (int off = 1; off <= 8; off <<= 1)
        ps += __shfl_xor(ps, off, 32);
      lrun[r] = lrun[r] * corr + ps;
      mrun[r] = mnew;
      o0[r] *= corr;  o1[r] *= corr;
      const int row = r + 8 * kh;
      P[wave][row][ 0 + ll] = (__bf16)p0;
      P[wave][row][16 + ll] = (__bf16)p1;
      P[wave][row][32 + ll] = (__bf16)p2;
      P[wave][row][48 + ll] = (__bf16)p3;
    }
    // ---- O += P * V : A-frag from own LDS tile (same-wave DS ordering),
    //      B-frag from Vt (d-major => contiguous m), 2 K-steps x 2 d-tiles
    #pragma unroll
    for (int kf = 0; kf < 2; ++kf) {
      const __bf16* Pr = &P[wave][ll][kf * 32 + kh * 8];
      v16bf aP = cat8(*(const v8bf*)Pr, *(const v8bf*)(Pr + 16));
      #pragma unroll
      for (int tc = 0; tc < 2; ++tc) {
        const __bf16* Vc = Vt + ((size_t)bh * DPAD + (tc * 16 + ll)) * LSEQ +
                           mt + kf * 32 + kh * 8;
        v16bf bf = cat8(*(const v8bf*)Vc, *(const v8bf*)(Vc + 16));
        if (tc == 0) o0 = wmma_bf16(aP, bf, o0);
        else         o1 = wmma_bf16(aP, bf, o1);
      }
    }
  }

  // ---- epilogue: normalize, store att in the raw-reshape flat layout
  //      attf[b][h*L*30 + l*30 + d]  (d=30,31 dropped)
  #pragma unroll
  for (int r = 0; r < 8; ++r) {
    const int lg = ltile + wave * 16 + r + 8 * kh;
    const float inv = 1.0f / lrun[r];
    const size_t base = ((size_t)bh * LSEQ + lg) * DTOT;
    attf[base + ll] = o0[r] * inv;
    if (ll < 14) attf[base + 16 + ll] = o1[r] * inv;
  }
}

// ---------------------------------------------------------------------------
// Kernel 3: output projection over the scrambled 300-channel view.
// One block per (b,l'); G[b,c,l'] = attf[b][c*L + l'] gathers the scramble.
// ---------------------------------------------------------------------------
__global__ __launch_bounds__(256) void msa_outproj(
    const float* __restrict__ attf, const float* __restrict__ wo,
    const float* __restrict__ bo, float* __restrict__ out)
{
  const int b = blockIdx.x >> 11, l = blockIdx.x & (LSEQ - 1);
  __shared__ float g[304];
  const float* fb = attf + (size_t)b * (NH * LSEQ * DTOT);
  for (int c = threadIdx.x; c < 300; c += 256)
    g[c] = fb[(size_t)c * LSEQ + l];
  __syncthreads();

  const int o = threadIdx.x;
  const float* w = wo + o * 300;
  float s = bo[o];
  #pragma unroll 5
  for (int c = 0; c < 300; c += 4) {
    float4 w4 = *(const float4*)(w + c);
    float4 g4 = *(const float4*)(g + c);
    s = fmaf(w4.x, g4.x, s); s = fmaf(w4.y, g4.y, s);
    s = fmaf(w4.z, g4.z, s); s = fmaf(w4.w, g4.w, s);
  }
  out[((size_t)b * NHID_ + o) * LSEQ + l] = s;
}

// ---------------------------------------------------------------------------
extern "C" void kernel_launch(void* const* d_in, const int* in_sizes, int n_in,
                              void* d_out, int out_size, void* d_ws, size_t ws_size,
                              hipStream_t stream) {
  (void)in_sizes; (void)n_in; (void)out_size; (void)ws_size;
  const float* x    = (const float*)d_in[0];
  const float* ax   = (const float*)d_in[1];
  const float* edge = (const float*)d_in[2];
  const float* wq   = (const float*)d_in[3];
  const float* bq   = (const float*)d_in[4];
  const float* wk   = (const float*)d_in[5];
  const float* bk   = (const float*)d_in[6];
  const float* wv   = (const float*)d_in[7];
  const float* bv   = (const float*)d_in[8];
  const float* wo   = (const float*)d_in[9];
  const float* bo   = (const float*)d_in[10];
  float* out = (float*)d_out;

  // Workspace: Kb | Vb | Vt (bf16, padded D=32) | attf (f32, D=30) = ~25.6 MB
  const size_t kvBytes = (size_t)BATCH * NH * LSEQ * DPAD * 2;  // 5,242,880
  char* ws = (char*)d_ws;
  __bf16* Kb   = (__bf16*)(ws);
  __bf16* Vb   = (__bf16*)(ws + kvBytes);
  __bf16* Vt   = (__bf16*)(ws + 2 * kvBytes);
  float*  attf = (float*) (ws + 3 * kvBytes);

  msa_proj<<<BATCH * LSEQ, 320, 0, stream>>>(x, ax, wq, bq, wk, bk, wv, bv,
                                             Kb, Vb, Vt);
  msa_flash<<<BATCH * NH * (LSEQ / 64), 128, 0, stream>>>(Kb, Vb, Vt, edge, attf);
  msa_outproj<<<BATCH * LSEQ, 256, 0, stream>>>(attf, wo, bo, out);
}
